// MultiHeadModel_11278584119317
// MI455X (gfx1250) — matrix operand order repaired and checked by reading, not compiled, not run on hardware
//
#include <hip/hip_runtime.h>

// MultiHeadModel fused kernel for MI455X (gfx1250, wave32, WMMA).
//
// Whole model collapses to one GEMM:  out = mask(route) * (x @ Wc + bc)[:,16:48]
// with Wc[128x48] = [W_lab | 0pad | W_enc@W_clf], route = argmax cols 0..7.
//
// Roofline: 256MB read + 64MB write @ 23.3 TB/s ~= 14us floor. bf16
// split-precision (hi*hi + lo*hi + hi*lo) WMMA_F32_16X16X32_BF16 with fp32
// accumulate keeps ~fp32 accuracy while staying under the bandwidth roofline.
//
// Round-2 restructure: R=2 row tiles + non-unrolled K loop + launch_bounds cap
// -> <=256 VGPRs (no s_set_vgpr_msb spam), ~5 waves/SIMD occupancy.
//
// Workspace: requires ws_size >= (128*48 + 48) * 4 = 24768 bytes.

#define N_ROWS    524288
#define D_IN      128
#define D_ENC     64
#define Y_DIM     8
#define S_DIM     4
#define C_COLS    48          // 0..7 labeler | 8..15 zero pad | 16..47 heads (i*4+s)
#define LDS_PITCH 52          // padded pitch (dwords): lanes 0-15 vs 16-31 disjoint banks
#define MACRO_ROWS 32
#define NMACRO    (N_ROWS / MACRO_ROWS)

typedef __attribute__((ext_vector_type(16))) __bf16       v16bf;
typedef __attribute__((ext_vector_type(8)))  float        v8f;
typedef __attribute__((ext_vector_type(8)))  unsigned int v8u;
typedef __attribute__((ext_vector_type(4)))  float        v4f;

static __device__ __forceinline__ unsigned pack2(__bf16 a, __bf16 b) {
  return (unsigned)__builtin_bit_cast(unsigned short, a) |
         ((unsigned)__builtin_bit_cast(unsigned short, b) << 16);
}

// ds_swizzle group-of-32 pure-xor: new_lane = lane ^ m  (imm = xor<<10 | and=0x1f)
#define SWZ_F(v, m) __builtin_bit_cast(float, \
    __builtin_amdgcn_ds_swizzle(__builtin_bit_cast(int, (v)), (((m) << 10) | 0x1f)))

// ---------------------------------------------------------------------------
// Prep: fuse encoder+classifier into Wc[128][48] and bc[48] (fp32, workspace).
// ---------------------------------------------------------------------------
__global__ void mhm_prep(const float* __restrict__ W_lab, const float* __restrict__ b_lab,
                         const float* __restrict__ W_enc, const float* __restrict__ b_enc,
                         const float* __restrict__ W_clf, const float* __restrict__ b_clf,
                         float* __restrict__ wc, float* __restrict__ bc) {
  const int c = blockIdx.x;    // 0..47
  const int k = threadIdx.x;   // 0..127
  float v;
  if (c < 8) {
    v = W_lab[k * Y_DIM + c];
  } else if (c < 16) {
    v = 0.0f;
  } else {
    const int i = (c - 16) >> 2, s = (c - 16) & 3;
    float acc = 0.0f;
    #pragma unroll 8
    for (int d = 0; d < D_ENC; ++d)
      acc = fmaf(W_enc[k * D_ENC + d], W_clf[(i * D_ENC + d) * S_DIM + s], acc);
    v = acc;
  }
  wc[k * C_COLS + c] = v;
  if (k == 0) {
    float b;
    if (c < 8)       b = b_lab[c];
    else if (c < 16) b = 0.0f;
    else {
      const int i = (c - 16) >> 2, s = (c - 16) & 3;
      float acc = b_clf[i * S_DIM + s];
      for (int d = 0; d < D_ENC; ++d)
        acc = fmaf(b_enc[d], W_clf[(i * D_ENC + d) * S_DIM + s], acc);
      b = acc;
    }
    bc[c] = b;
  }
}

// ---------------------------------------------------------------------------
// Main: bandwidth-bound streaming GEMM + route/mask epilogue.
// Wave owns 32 rows (2 x 16-row tiles). K loop NOT unrolled so only one
// K-step's B operands (3 tiles x hi/lo = 48 VGPRs) are live at a time;
// 6 x v8f fp32 accumulators (48 VGPRs) persist across K.
// ---------------------------------------------------------------------------
__global__ __launch_bounds__(256, 4)
void mhm_main(const float* __restrict__ x, const float* __restrict__ wc,
              const float* __restrict__ bc, float* __restrict__ out) {
  __shared__ unsigned int lds_hi[64 * LDS_PITCH];   // Wc hi-bf16, K-pair packed
  __shared__ unsigned int lds_lo[64 * LDS_PITCH];   // Wc lo-bf16 residual

  // Stage Wc -> LDS as packed bf16 (hi) + bf16 residual (lo). kk = K/2.
  for (int idx = threadIdx.x; idx < 64 * C_COLS; idx += 256) {
    const int kk = idx / C_COLS, c = idx % C_COLS;
    const float f0 = wc[(2 * kk) * C_COLS + c];
    const float f1 = wc[(2 * kk + 1) * C_COLS + c];
    const __bf16 h0 = (__bf16)f0, h1 = (__bf16)f1;
    const __bf16 l0 = (__bf16)(f0 - (float)h0);
    const __bf16 l1 = (__bf16)(f1 - (float)h1);
    lds_hi[kk * LDS_PITCH + c] = pack2(h0, h1);
    lds_lo[kk * LDS_PITCH + c] = pack2(l0, l1);
  }
  __syncthreads();

  const int lane   = threadIdx.x & 31;
  const int lane15 = lane & 15;
  const int half   = lane >> 4;              // 0: K-lo / rows 0-7, 1: K-hi / rows 8-15

  const float bias0 = bc[lane15];            // logits bias (0 for pad cols)
  const float bias1 = bc[16 + lane15];       // heads cols 0..15
  const float bias2 = bc[32 + lane15];       // heads cols 16..31

  const int waveId = (blockIdx.x * blockDim.x + threadIdx.x) >> 5;
  const int nWaves = (gridDim.x * blockDim.x) >> 5;

  for (int m = waveId; m < NMACRO; m += nWaves) {
    const int rowbase = m * MACRO_ROWS;

    if (m + nWaves < NMACRO) {  // wave-uniform; speculative stream prefetch
      __builtin_prefetch(x + (size_t)(m + nWaves) * MACRO_ROWS * D_IN
                           + (size_t)lane * (MACRO_ROWS * D_IN / 32), 0, 0);
    }

    v8f C[3][2] = {};           // [tile][row-tile] fp32 accumulators (48 VGPRs)

    // Per-lane A base: row = rowbase + rt*16 + (lane&15); K offset half*16.
    const float* xb0 = x + (size_t)(rowbase + lane15) * D_IN + half * 16;
    const float* xb1 = xb0 + 16 * D_IN;

    #pragma unroll 1
    for (int k0 = 0; k0 < 4; ++k0) {          // K-step of 32, kept as a real loop
      // ---- B operands for this K-step (only 48 VGPRs live) ----
      const unsigned* lh = &lds_hi[(k0 * 16 + half * 8) * LDS_PITCH + lane15];
      const unsigned* ll = &lds_lo[(k0 * 16 + half * 8) * LDS_PITCH + lane15];
      v8u bhv[3], blv[3];
      #pragma unroll
      for (int t = 0; t < 3; ++t) {
        #pragma unroll
        for (int j = 0; j < 8; ++j) {
          bhv[t][j] = lh[j * LDS_PITCH + t * 16];
          blv[t][j] = ll[j * LDS_PITCH + t * 16];
        }
      }
      const v16bf Bh0 = __builtin_bit_cast(v16bf, bhv[0]);
      const v16bf Bh1 = __builtin_bit_cast(v16bf, bhv[1]);
      const v16bf Bh2 = __builtin_bit_cast(v16bf, bhv[2]);
      const v16bf Bl0 = __builtin_bit_cast(v16bf, blv[0]);
      const v16bf Bl1 = __builtin_bit_cast(v16bf, blv[1]);
      const v16bf Bl2 = __builtin_bit_cast(v16bf, blv[2]);

      #pragma unroll
      for (int rt = 0; rt < 2; ++rt) {
        // A 16x32 bf16 layout: lane holds row (lane&15); elems 0..15 = K+0..15
        // (K = k0*32 + half*16). One contiguous 64B run.
        const v4f* xr4 = reinterpret_cast<const v4f*>((rt ? xb1 : xb0) + k0 * 32);
        v4f av[4];
        av[0] = xr4[0]; av[1] = xr4[1];   // run 1
        av[2] = xr4[2]; av[3] = xr4[3];   // run 2

        v16bf Ah, Al;
        #pragma unroll
        for (int q = 0; q < 4; ++q) {
          #pragma unroll
          for (int e = 0; e < 4; ++e) {
            const float fx = av[q][e];
            const __bf16 h = (__bf16)fx;
            Ah[q * 4 + e] = h;
            Al[q * 4 + e] = (__bf16)(fx - (float)h);
          }
        }

        // Split-precision: hi*hi + lo*hi + hi*lo  (lo*lo negligible)
        C[0][rt] = __builtin_amdgcn_wmma_f32_16x16x32_bf16(false, Ah, false, Bh0, (short)0, C[0][rt], false, false);
        C[0][rt] = __builtin_amdgcn_wmma_f32_16x16x32_bf16(false, Al, false, Bh0, (short)0, C[0][rt], false, false);
        C[0][rt] = __builtin_amdgcn_wmma_f32_16x16x32_bf16(false, Ah, false, Bl0, (short)0, C[0][rt], false, false);

        C[1][rt] = __builtin_amdgcn_wmma_f32_16x16x32_bf16(false, Ah, false, Bh1, (short)0, C[1][rt], false, false);
        C[1][rt] = __builtin_amdgcn_wmma_f32_16x16x32_bf16(false, Al, false, Bh1, (short)0, C[1][rt], false, false);
        C[1][rt] = __builtin_amdgcn_wmma_f32_16x16x32_bf16(false, Ah, false, Bl1, (short)0, C[1][rt], false, false);

        C[2][rt] = __builtin_amdgcn_wmma_f32_16x16x32_bf16(false, Ah, false, Bh2, (short)0, C[2][rt], false, false);
        C[2][rt] = __builtin_amdgcn_wmma_f32_16x16x32_bf16(false, Al, false, Bh2, (short)0, C[2][rt], false, false);
        C[2][rt] = __builtin_amdgcn_wmma_f32_16x16x32_bf16(false, Ah, false, Bl2, (short)0, C[2][rt], false, false);
      }
    }

    // Epilogue: argmax of logits (C-tile 0, cols 0..7) then mask + store.
    // C layout: VGPR r, lanes 0-15 -> row r, N=lane; lanes 16-31 -> row 8+r.
    #pragma unroll
    for (int rt = 0; rt < 2; ++rt) {
      const int ihead = lane15 >> 2;   // head index of this lane's tile-1 col
      float* pb = out + (size_t)(rowbase + rt * 16 + half * 8) * 32 + lane15;

      #pragma unroll
      for (int r = 0; r < 8; ++r) {
        // ---- route for row (r + 8*half) ----
        const float ov = (lane15 < 8) ? (C[0][rt][r] + bias0) : -3.402823466e+38f;
        float mv = ov;
        mv = fmaxf(mv, SWZ_F(mv, 1));
        mv = fmaxf(mv, SWZ_F(mv, 2));
        mv = fmaxf(mv, SWZ_F(mv, 4));
        mv = fmaxf(mv, SWZ_F(mv, 8));   // all 16 lanes of the half hold the row max
        const unsigned bal = __builtin_amdgcn_ballot_w32((lane15 < 8) && (ov == mv));
        const unsigned grp = (bal >> (half * 16)) & 0xffu;
        const int route = (int)__builtin_ctz(grp | 0x100u);  // lowest index on ties

        // ---- masked heads, cols 0..15 (tile1) and 16..31 (tile2) ----
        float h1 = C[1][rt][r] + bias1;
        pb[r * 32]      = (route == ihead)     ? h1 : 0.0f;
        float h2 = C[2][rt][r] + bias2;
        pb[r * 32 + 16] = (route == ihead + 4) ? h2 : 0.0f;
      }
    }
  }
}

// ---------------------------------------------------------------------------
extern "C" void kernel_launch(void* const* d_in, const int* in_sizes, int n_in,
                              void* d_out, int out_size, void* d_ws, size_t ws_size,
                              hipStream_t stream) {
  const float* x     = (const float*)d_in[0];
  const float* W_lab = (const float*)d_in[1];
  const float* b_lab = (const float*)d_in[2];
  const float* W_enc = (const float*)d_in[3];
  const float* b_enc = (const float*)d_in[4];
  const float* W_clf = (const float*)d_in[5];
  const float* b_clf = (const float*)d_in[6];
  float* out = (float*)d_out;

  float* wc = (float*)d_ws;                 // [128][48]
  float* bc = wc + D_IN * C_COLS;           // [48]
  (void)in_sizes; (void)n_in; (void)out_size; (void)ws_size;

  mhm_prep<<<dim3(C_COLS), dim3(D_IN), 0, stream>>>(W_lab, b_lab, W_enc, b_enc,
                                                    W_clf, b_clf, wc, bc);
  // 1024 WGs x 8 waves = 8192 waves; 16384 macro-tiles -> 2 per wave.
  mhm_main<<<dim3(1024), dim3(256), 0, stream>>>(x, wc, bc, out);
}
